// NEURONSv2_61950608277920
// MI455X (gfx1250) — matrix operand order repaired and checked by reading, not compiled, not run on hardware
//
#include <hip/hip_runtime.h>

#define DEV __device__ __forceinline__

typedef __attribute__((ext_vector_type(16))) __bf16 v16bf;
typedef __attribute__((ext_vector_type(8)))  __bf16 v8bf;
typedef __attribute__((ext_vector_type(8)))  float  v8f;

constexpr int   D       = 1024;
constexpr int   Bb      = 64;
constexpr int   T       = 128;
constexpr int   NTILES  = 64;   // D/16
constexpr int   KTILES  = 32;   // D/32
constexpr size_t FRAG_MAT = (size_t)NTILES * KTILES * 32 * 16; // elems per matrix (=D*D)

constexpr float DDEC   = 0.9048374180359595f;  // exp(-1/10) mem & dend decay
constexpr float SDEC   = 0.8187307530779818f;  // exp(-1/5) syn decay
constexpr float OMEGA  = 0.06283185307179587f;
constexpr float TWO_PI = 6.283185307179586f;
constexpr float INV2PI = 0.15915494309189535f;

DEV v8f v8zero() { v8f z = {0.f,0.f,0.f,0.f,0.f,0.f,0.f,0.f}; return z; }

DEV v8f wmma_bf16(v16bf a, v16bf b, v8f c) {
  return __builtin_amdgcn_wmma_f32_16x16x32_bf16(false, a, false, b, (short)0, c, false, false);
}

// A-fragment (16x32 bf16 tile) per ISA layout: lane row = lane&15,
// lanes 0-15 hold K {0..7,16..23}, lanes 16-31 hold K {8..15,24..31}.
DEV v16bf load_a_f32(const float* __restrict__ row0, int kt, int half8) {
  const float* p = row0 + kt * 32 + half8;
  v16bf a;
#pragma unroll
  for (int i = 0; i < 8; i++) a[i]     = (__bf16)p[i];
#pragma unroll
  for (int i = 0; i < 8; i++) a[8 + i] = (__bf16)p[16 + i];
  return a;
}

DEV v16bf load_a_lds(const __bf16* row0, int kt, int half8) {
  const __bf16* p = row0 + kt * 32 + half8;
  v8bf lo = *(const v8bf*)p;
  v8bf hi = *(const v8bf*)(p + 16);
  v16bf a;
#pragma unroll
  for (int i = 0; i < 8; i++) { a[i] = lo[i]; a[8 + i] = hi[i]; }
  return a;
}

// ---------------------------------------------------------------------------
// Pre-swizzle all 14 weight matrices fp32 -> bf16 B-fragments.
// Fragment order: [mat][ntile][ktile][lane][16 bf16]; B element (k,n) = W[n][k].
// Lane n = lane&15; lanes 0-15 carry K {0..7,16..23} of the 32-K tile, lanes 16-31 carry {8..15,24..31}.
// Mats: 0-3 Wb1, 4-5 Wa1, 6 W1, 7-10 Wb2, 11-12 Wa2, 13 W2.
// ---------------------------------------------------------------------------
__global__ __launch_bounds__(256) void convert_weights(
    const float* __restrict__ Wb1, const float* __restrict__ Wa1, const float* __restrict__ W1,
    const float* __restrict__ Wb2, const float* __restrict__ Wa2, const float* __restrict__ W2,
    __bf16* __restrict__ frag)
{
  int gid = blockIdx.x * 256 + threadIdx.x;
  const int total = 14 * NTILES * KTILES * 32;
  if (gid >= total) return;
  int lane = gid & 31;
  int kt   = (gid >> 5)  & (KTILES - 1);
  int nt   = (gid >> 10) & (NTILES - 1);
  int mat  = gid >> 16;
  const float* src;
  if      (mat < 4)  src = Wb1 + (size_t)mat * D * D;
  else if (mat < 6)  src = Wa1 + (size_t)(mat - 4) * D * D;
  else if (mat == 6) src = W1;
  else if (mat < 11) src = Wb2 + (size_t)(mat - 7) * D * D;
  else if (mat < 13) src = Wa2 + (size_t)(mat - 11) * D * D;
  else               src = W2;
  int n  = nt * 16 + (lane & 15);
  int k0 = kt * 32 + ((lane & 16) ? 8 : 0);
  const float* s = src + (size_t)n * D + k0;
  v8bf lo, hi;
#pragma unroll
  for (int i = 0; i < 8; i++) lo[i] = (__bf16)s[i];
#pragma unroll
  for (int i = 0; i < 8; i++) hi[i] = (__bf16)s[16 + i];
  v8bf* dst = (v8bf*)(frag + (size_t)gid * 16);
  dst[0] = lo; dst[1] = hi;
}

// ---------------------------------------------------------------------------
// Parallel branch GEMMs: U[row=t*64+b][n] = 0.25 * sum over 6 mats relu(X W_m^T).
// Block = one 16-row A tile (staged once in LDS as bf16, shared by 8 waves),
// wave = one 16x16 output tile across all 6 matrices (K fully in-wave).
// Input addressed as in[b*bstr + t*tstr + k] (handles (B,T,D) x and (T,B,D) s1).
// ---------------------------------------------------------------------------
__global__ __launch_bounds__(256) void branch_gemm(
    const float* __restrict__ in, long bstr, long tstr,
    const __bf16* __restrict__ frag,     // 6 matrices, pre-swizzled
    float* __restrict__ U)
{
  __shared__ __bf16 xs[16 * D];          // A tile (rows of this rt), bf16

  const int tid   = threadIdx.x;
  const int wave  = tid >> 5, lane = tid & 31;
  const int l15   = lane & 15;
  const int half8 = (lane & 16) ? 8 : 0;
  const int rt    = blockIdx.x >> 3;                 // 0..511  (row tile)
  const int nt    = (blockIdx.x & 7) * 8 + wave;     // 0..63   (col tile)

  // --- cooperative A stage: 16 rows x 1024 cols, fp32 -> bf16 ---
  {
    const int r  = tid >> 4;            // 0..15
    const int k0 = (tid & 15) * 64;     // 64 contiguous k per thread
    const int arow = rt * 16 + r;
    const int t = arow >> 6, b = arow & 63;
    const float* src = in + (size_t)b * bstr + (size_t)t * tstr + k0;
#pragma unroll
    for (int i = 0; i < 64; i += 8) {
      float4 f0 = *(const float4*)(src + i);
      float4 f1 = *(const float4*)(src + i + 4);
      v8bf v;
      v[0] = (__bf16)f0.x; v[1] = (__bf16)f0.y; v[2] = (__bf16)f0.z; v[3] = (__bf16)f0.w;
      v[4] = (__bf16)f1.x; v[5] = (__bf16)f1.y; v[6] = (__bf16)f1.z; v[7] = (__bf16)f1.w;
      *(v8bf*)(xs + r * D + k0 + i) = v;
    }
  }
  __syncthreads();

  v8f acc[6];
#pragma unroll
  for (int m = 0; m < 6; m++) acc[m] = v8zero();
  const v16bf* fb = (const v16bf*)frag;
  const __bf16* arow = xs + l15 * D;

  for (int kt = 0; kt < KTILES; ++kt) {
    v16bf a = load_a_lds(arow, kt, half8);           // A shared by all 6 matrices
#pragma unroll
    for (int m = 0; m < 6; m++) {
      size_t idx = ((size_t)(m * NTILES + nt) * KTILES + kt) * 32 + lane;
      if (kt + 1 < KTILES)                           // pull next k-tile toward L0
        __builtin_prefetch((const void*)(fb + idx + 32), 0, 1);
      acc[m] = wmma_bf16(a, fb[idx], acc[m]);
    }
  }

  const int n = nt * 16 + l15;
#pragma unroll
  for (int v = 0; v < 8; v++) {
    float u = 0.f;
#pragma unroll
    for (int m = 0; m < 6; m++) u += fmaxf(acc[m][v], 0.f);
    U[(size_t)(rt * 16 + half8 + v) * D + n] = 0.25f * u;  // basal/4 + 0.5*apical/2
  }
}

// ---------------------------------------------------------------------------
// Sequential recurrence. grid = 4 workgroups (16 batch rows each), 1024 thr.
// dend state in LDS (A-matrix source); syn/mem/phase in VGPRs mapped exactly
// to the WMMA C/D layout so the GEMM result feeds LIF with no staging.
// ---------------------------------------------------------------------------
template<int USE_EXT>
__global__ __launch_bounds__(1024) void recurrent(
    const float* __restrict__ U,         // precomputed branch drive, (T,B,D)
    const float* __restrict__ ext,       // p1 (layer 2 only), (T,B,D)
    const __bf16* __restrict__ wfrag,    // W-matrix fragments
    const float* __restrict__ bias,
    float* __restrict__ out_spk,         // s1 (layer 1)
    float* __restrict__ out_ph,          // p1 (layer 1)
    float* __restrict__ out_rate)        // final (B,D) (layer 2)
{
  __shared__ float dendb[16 * D];
  __shared__ float redS[16][32];
  __shared__ float redC[16][32];
  __shared__ float rmS[16];
  __shared__ float rmC[16];

  const int tid = threadIdx.x;
  const int wave = tid >> 5, lane = tid & 31, l15 = lane & 15;
  const int half8 = (lane & 16) ? 8 : 0;
  const int bg = blockIdx.x;
  const int er = tid >> 6, en0 = (tid & 63) * 16;   // elementwise mapping for U->dend

#pragma unroll
  for (int i = 0; i < 16; i++) dendb[tid * 16 + i] = 0.f;

  float syn[2][8], mem[2][8], ph[2][8], rate[2][8];
#pragma unroll
  for (int j = 0; j < 2; j++)
#pragma unroll
    for (int v = 0; v < 8; v++) { syn[j][v] = 0.f; mem[j][v] = 0.f; ph[j][v] = 0.f; rate[j][v] = 0.f; }

  float bval[2];
#pragma unroll
  for (int j = 0; j < 2; j++) bval[j] = bias[(wave + j * 32) * 16 + l15];

  const v16bf* fb = (const v16bf*)wfrag;
  __syncthreads();

  for (int t = 0; t < T; t++) {
    // --- dendrite recurrence: dend = dd*dend + U[t] ---
    const float* up = U + (size_t)t * (Bb * D) + (size_t)(bg * 16 + er) * D + en0;
#pragma unroll
    for (int i = 0; i < 16; i++) {
      int idx = er * D + en0 + i;
      dendb[idx] = DDEC * dendb[idx] + up[i];
    }
    __syncthreads();

    // --- y = dend @ W^T : wave owns ntiles {wave, wave+32}, shared A load ---
    v8f acc0 = v8zero(), acc1 = v8zero();
    {
      const float* arow = dendb + l15 * D;          // A row = local batch row
      const int nt0 = wave, nt1 = wave + 32;
      for (int kt = 0; kt < KTILES; ++kt) {
        v16bf a = load_a_f32(arow, kt, half8);
        size_t i0 = ((size_t)nt0 * KTILES + kt) * 32 + lane;
        size_t i1 = ((size_t)nt1 * KTILES + kt) * 32 + lane;
        if (kt + 1 < KTILES) {
          __builtin_prefetch((const void*)(fb + i0 + 32), 0, 1);
          __builtin_prefetch((const void*)(fb + i1 + 32), 0, 1);
        }
        acc0 = wmma_bf16(a, fb[i0], acc0);
        acc1 = wmma_bf16(a, fb[i1], acc1);
      }
    }
    v8f y[2] = {acc0, acc1};
    __syncthreads();  // all dend reads done before next step's writes

    // --- Kuramoto mean-field row reduction (layer 1 only) ---
    if (!USE_EXT) {
      float ss[8], cc[8];
#pragma unroll
      for (int v = 0; v < 8; v++) { ss[v] = 0.f; cc[v] = 0.f; }
#pragma unroll
      for (int j = 0; j < 2; j++)
#pragma unroll
        for (int v = 0; v < 8; v++) { ss[v] += __sinf(ph[j][v]); cc[v] += __cosf(ph[j][v]); }
      // butterfly over the 16-lane half (masks < 16 keep halves separate)
#pragma unroll
      for (int m = 1; m < 16; m <<= 1) {
#pragma unroll
        for (int v = 0; v < 8; v++) {
          ss[v] += __shfl_xor(ss[v], m, 32);
          cc[v] += __shfl_xor(cc[v], m, 32);
        }
      }
      if (l15 == 0) {
#pragma unroll
        for (int v = 0; v < 8; v++) { redS[half8 + v][wave] = ss[v]; redC[half8 + v][wave] = cc[v]; }
      }
      __syncthreads();
      if (tid < 16) {
        float s = 0.f, c = 0.f;
#pragma unroll
        for (int w = 0; w < 32; w++) { s += redS[tid][w]; c += redC[tid][w]; }
        rmS[tid] = s * (1.f / D); rmC[tid] = c * (1.f / D);
      }
      __syncthreads();
    }

    // --- LIF + phase update, state mapped to WMMA C/D layout ---
#pragma unroll
    for (int j = 0; j < 2; j++) {
      const int n = (wave + j * 32) * 16 + l15;
#pragma unroll
      for (int v = 0; v < 8; v++) {
        const int M = half8 + v;
        float sy = SDEC * syn[j][v] + y[j][v] + bval[j];
        float mm = DDEC * mem[j][v] + sy;
        float spike = (mm > 1.f) ? 1.f : 0.f;
        mm -= spike;
        float p = ph[j][v];
        float drive;
        if (USE_EXT) {
          float e = ext[(size_t)t * (Bb * D) + (size_t)(bg * 16 + M) * D + n];
          drive = 0.5f * __sinf(e - p);
        } else {
          drive = 0.5f * (__cosf(p) * rmS[M] - __sinf(p) * rmC[M]);
        }
        p += OMEGA + drive + spike;
        p -= floorf(p * INV2PI) * TWO_PI;
        syn[j][v] = sy; mem[j][v] = mm; ph[j][v] = p;
        if (!USE_EXT) {
          size_t o = (size_t)t * (Bb * D) + (size_t)(bg * 16 + M) * D + n;
          out_spk[o] = spike;
          out_ph[o]  = p;
        } else {
          rate[j][v] += spike;
        }
      }
    }
  }

  if (USE_EXT) {
#pragma unroll
    for (int j = 0; j < 2; j++) {
      const int n = (wave + j * 32) * 16 + l15;
#pragma unroll
      for (int v = 0; v < 8; v++)
        out_rate[(size_t)(bg * 16 + half8 + v) * D + n] = rate[j][v] * (1.f / T);
    }
  }
}

// ---------------------------------------------------------------------------
extern "C" void kernel_launch(void* const* d_in, const int* in_sizes, int n_in,
                              void* d_out, int out_size, void* d_ws, size_t ws_size,
                              hipStream_t stream) {
  const float* x   = (const float*)d_in[0];
  const float* W1  = (const float*)d_in[1];
  const float* b1  = (const float*)d_in[2];
  const float* Wb1 = (const float*)d_in[3];
  const float* Wa1 = (const float*)d_in[4];
  const float* W2  = (const float*)d_in[5];
  const float* b2  = (const float*)d_in[6];
  const float* Wb2 = (const float*)d_in[7];
  const float* Wa2 = (const float*)d_in[8];
  float* out = (float*)d_out;

  char* ws = (char*)d_ws;
  __bf16* frag = (__bf16*)ws;                         // 28 MB: 14 swizzled bf16 matrices
  float* U  = (float*)(ws + 28ull * 1024 * 1024);     // 32 MB: branch drive (reused per layer)
  float* s1 = (float*)(ws + 60ull * 1024 * 1024);     // 32 MB: layer-1 spikes (T,B,D)
  float* p1 = (float*)(ws + 92ull * 1024 * 1024);     // 32 MB: layer-1 phases (T,B,D)

  {
    int total = 14 * NTILES * KTILES * 32;
    convert_weights<<<(total + 255) / 256, 256, 0, stream>>>(Wb1, Wa1, W1, Wb2, Wa2, W2, frag);
  }
  // Layer 1: x is (B,T,D) -> bstr=T*D, tstr=D
  branch_gemm<<<4096, 256, 0, stream>>>(x, (long)(T * D), (long)D, frag, U);
  recurrent<0><<<4, 1024, 0, stream>>>(U, p1, frag + 6 * FRAG_MAT, b1, s1, p1, out);
  // Layer 2: s1 is (T,B,D) -> bstr=D, tstr=B*D
  branch_gemm<<<4096, 256, 0, stream>>>(s1, (long)D, (long)(Bb * D), frag + 7 * FRAG_MAT, U);
  recurrent<1><<<4, 1024, 0, stream>>>(U, p1, frag + 13 * FRAG_MAT, b2, s1, s1, out);
}
